// Res_EGblk_1_687194767630
// MI455X (gfx1250) — matrix operand order repaired and checked by reading, not compiled, not run on hardware
//
#include <hip/hip_runtime.h>
#include <hip/hip_fp16.h>

#define N_NODES 50000
#define N_EDGES 800000
#define ND 96
#define ED 64
#define EFD 32
#define CATD (ND + ED + EFD)   // 192
#define BN_EPS 1e-5f
#define EDGE_WAVES 8
#define NG_WAVES 8

typedef _Float16 v16h __attribute__((ext_vector_type(16)));
typedef float    v8f  __attribute__((ext_vector_type(8)));

static __device__ __forceinline__ v8f wmma_f16(v16h a, v16h b, v8f c) {
  // D = A(16x32 f16) * B(32x16 f16) + C(16x16 f32)
  return __builtin_amdgcn_wmma_f32_16x16x32_f16(false, a, false, b, (short)0, c,
                                                false, false);
}

// ---------------------------------------------------------------------------
// Convert weight matrices to f16 once (they are tiny; stay in L2 afterwards).
// ---------------------------------------------------------------------------
__global__ void convert_weights_kernel(const float* __restrict__ Wn,
                                       const float* __restrict__ Wec,
                                       const float* __restrict__ Wg,
                                       _Float16* __restrict__ hWn,
                                       _Float16* __restrict__ hWec,
                                       _Float16* __restrict__ hWg) {
  int i = blockIdx.x * blockDim.x + threadIdx.x;
  if (i < ND * ND) {
    hWn[i] = (_Float16)Wn[i];
    hWg[i] = (_Float16)Wg[i];
  }
  if (i < ED * CATD) hWec[i] = (_Float16)Wec[i];
}

// ---------------------------------------------------------------------------
// Fused edge kernel: per wave, one 16-edge tile.
//   GEMM1: m = (x[src]-x[dst]) @ Wn^T   (16x96x96, f16 WMMA)
//   GEMM2: ef = [m | edge_attr | edge_f] @ Wec^T (16x64x192, f16 WMMA)
//   head : w = relu(relu(ef) @ Ww^T + bw), atomicAdd into deg[dst]
// Raw (pre-BN) edge_feat is written straight into the d_out edge region.
// ---------------------------------------------------------------------------
__global__ void __launch_bounds__(256) edge_kernel(
    const float* __restrict__ x, const long long* __restrict__ ei,
    const float* __restrict__ edge_attr, const float* __restrict__ edge_f,
    const _Float16* __restrict__ hWn, const _Float16* __restrict__ hWec,
    const float* __restrict__ bn_, const float* __restrict__ bec,
    const float* __restrict__ Ww, const float* __restrict__ bw,
    float* __restrict__ ef_raw, float* __restrict__ w_out,
    float* __restrict__ deg) {
  __shared__ __align__(32) _Float16 sWn[ND * ND];      // 18 KB
  __shared__ __align__(32) _Float16 sWec[ED * CATD];   // 24 KB
  __shared__ __align__(32) _Float16 sM[EDGE_WAVES][16 * ND];  // 24 KB

  {  // cooperative weight staging (f16, 128-bit copies)
    const uint4* s0 = (const uint4*)hWn;
    uint4* d0 = (uint4*)sWn;
    for (int i = threadIdx.x; i < (ND * ND) / 8; i += blockDim.x) d0[i] = s0[i];
    const uint4* s1 = (const uint4*)hWec;
    uint4* d1 = (uint4*)sWec;
    for (int i = threadIdx.x; i < (ED * CATD) / 8; i += blockDim.x) d1[i] = s1[i];
  }
  __syncthreads();

  const int wave = threadIdx.x >> 5;
  const int lane = threadIdx.x & 31;
  const int tile = blockIdx.x * EDGE_WAVES + wave;  // exactly 50000 tiles
  const int e0 = tile * 16;

  // wave32 fragment lane mappings (k order used here: contiguous, k = 16*h + i)
  const int am  = lane & 15;              // A: row index
  const int hk  = (lane < 16) ? 0 : 16;   // A: K base
  const int bn  = lane & 15;              // B: column index
  const int bkb = (lane < 16) ? 0 : 16;   // B: K base
  const int dn  = lane & 15;              // D: column index
  const int dmb = (lane < 16) ? 0 : 8;    // D: row base (8 rows in slots)

  const long long es = ei[e0 + am];
  const long long ed = ei[N_EDGES + e0 + am];
  const float* xs = x + (size_t)es * ND;
  const float* xd = x + (size_t)ed * ND;

  // ---- GEMM1 A fragments: diff = x_src - x_dst, converted to f16 ----
  v16h a1[3];
#pragma unroll
  for (int kc = 0; kc < 3; ++kc)
#pragma unroll
    for (int i = 0; i < 16; ++i) {
      int k = kc * 32 + hk + i;
      a1[kc][i] = (_Float16)(xs[k] - xd[k]);
    }

  _Float16* mrow = &sM[wave][0];
#pragma unroll
  for (int nt = 0; nt < 6; ++nt) {
    v8f c;
#pragma unroll
    for (int r = 0; r < 8; ++r) c[r] = 0.f;
#pragma unroll
    for (int kc = 0; kc < 3; ++kc) {
      const v16h b = *(const v16h*)&sWn[(nt * 16 + bn) * ND + kc * 32 + bkb];
      c = wmma_f16(a1[kc], b, c);
    }
    const float bias = bn_[nt * 16 + dn];
#pragma unroll
    for (int r = 0; r < 8; ++r)  // D-layout -> row-major f16 staging tile
      mrow[(dmb + r) * ND + nt * 16 + dn] = (_Float16)(c[r] + bias);
  }
  __syncthreads();  // sM visible (block-uniform barrier)

  // ---- GEMM2: A = [m | edge_attr | edge_f], K = 192 ----
  v8f acc2[4];
#pragma unroll
  for (int nt = 0; nt < 4; ++nt)
#pragma unroll
    for (int r = 0; r < 8; ++r) acc2[nt][r] = 0.f;

  const float* ea_row = edge_attr + (size_t)(e0 + am) * ED;
  const float* ef_row = edge_f + (size_t)(e0 + am) * EFD;

#pragma unroll
  for (int kc = 0; kc < 6; ++kc) {
    v16h a2;
#pragma unroll
    for (int i = 0; i < 16; ++i) {
      int k = kc * 32 + hk + i;
      _Float16 v;
      if (k < ND)            v = mrow[am * ND + k];
      else if (k < ND + ED)  v = (_Float16)ea_row[k - ND];
      else                   v = (_Float16)ef_row[k - ND - ED];
      a2[i] = v;
    }
#pragma unroll
    for (int nt = 0; nt < 4; ++nt) {
      const v16h b = *(const v16h*)&sWec[(nt * 16 + bn) * CATD + kc * 32 + bkb];
      acc2[nt] = wmma_f16(a2, b, acc2[nt]);
    }
  }

  // ---- epilogue: store raw edge_feat, compute edge-weight head ----
  float wacc[8];
#pragma unroll
  for (int r = 0; r < 8; ++r) wacc[r] = 0.f;
#pragma unroll
  for (int nt = 0; nt < 4; ++nt) {
    const int f = nt * 16 + dn;
    const float bias = bec[f];
    const float wwf = Ww[f];
#pragma unroll
    for (int r = 0; r < 8; ++r) {
      float v = acc2[nt][r] + bias;
      ef_raw[(size_t)(e0 + dmb + r) * ED + f] = v;  // pre-BN edge_feat
      wacc[r] += fmaxf(v, 0.f) * wwf;
    }
  }
  const float bw0 = bw[0];
#pragma unroll
  for (int r = 0; r < 8; ++r) {  // reduce across the 16 columns in each half
    float s = wacc[r];
    for (int off = 1; off < 16; off <<= 1) s += __shfl_xor(s, off, 32);
    wacc[r] = s;
  }
  if (dn == 0) {  // lane 0 -> rows 0..7, lane 16 -> rows 8..15
#pragma unroll
    for (int r = 0; r < 8; ++r) {
      const int e = e0 + dmb + r;
      const float w = fmaxf(wacc[r] + bw0, 0.f);
      w_out[e] = w;
      const long long d = ei[N_EDGES + e];
      atomicAdd(&deg[d], w);
    }
  }
}

// ---------------------------------------------------------------------------
// xw = x @ Wg^T  (50000 x 96 x 96, f16 WMMA)
// ---------------------------------------------------------------------------
__global__ void __launch_bounds__(256) node_gemm_kernel(
    const float* __restrict__ x, const _Float16* __restrict__ hWg,
    float* __restrict__ xw) {
  __shared__ __align__(32) _Float16 sWg[ND * ND];
  {
    const uint4* s0 = (const uint4*)hWg;
    uint4* d0 = (uint4*)sWg;
    for (int i = threadIdx.x; i < (ND * ND) / 8; i += blockDim.x) d0[i] = s0[i];
  }
  __syncthreads();

  const int wave = threadIdx.x >> 5;
  const int lane = threadIdx.x & 31;
  const int tile = blockIdx.x * NG_WAVES + wave;
  if (tile >= N_NODES / 16) return;  // wave-uniform: EXEC stays full for WMMA
  const int n0 = tile * 16;

  const int am = lane & 15, hk = (lane < 16) ? 0 : 16;
  const int bn = lane & 15, bkb = (lane < 16) ? 0 : 16;
  const int dn = lane & 15, dmb = (lane < 16) ? 0 : 8;
  const float* xr = x + (size_t)(n0 + am) * ND;

  v16h a[3];
#pragma unroll
  for (int kc = 0; kc < 3; ++kc)
#pragma unroll
    for (int i = 0; i < 16; ++i) {
      int k = kc * 32 + hk + i;
      a[kc][i] = (_Float16)xr[k];
    }
#pragma unroll
  for (int nt = 0; nt < 6; ++nt) {
    v8f c;
#pragma unroll
    for (int r = 0; r < 8; ++r) c[r] = 0.f;
#pragma unroll
    for (int kc = 0; kc < 3; ++kc) {
      const v16h b = *(const v16h*)&sWg[(nt * 16 + bn) * ND + kc * 32 + bkb];
      c = wmma_f16(a[kc], b, c);
    }
#pragma unroll
    for (int r = 0; r < 8; ++r)
      xw[(size_t)(n0 + dmb + r) * ND + nt * 16 + dn] = c[r];
  }
}

__global__ void dinv_kernel(const float* __restrict__ deg,
                            float* __restrict__ dinv) {
  int i = blockIdx.x * blockDim.x + threadIdx.x;
  if (i < N_NODES) {
    float d = deg[i];
    dinv[i] = (d > 0.f) ? rsqrtf(d) : 0.f;
  }
}

// agg[dst] += dinv[src]*w*dinv[dst] * xw[src]   (one thread per edge)
__global__ void scatter_kernel(const long long* __restrict__ ei,
                               const float* __restrict__ w,
                               const float* __restrict__ dinv,
                               const float* __restrict__ xw,
                               float* __restrict__ agg) {
  int e = blockIdx.x * blockDim.x + threadIdx.x;
  if (e >= N_EDGES) return;
  const long long s = ei[e];
  const long long d = ei[N_EDGES + e];
  const float norm = dinv[s] * w[e] * dinv[d];
  const float4* xs = (const float4*)(xw + (size_t)s * ND);
  __builtin_prefetch(xs, 0, 0);
  float* ar = agg + (size_t)d * ND;
#pragma unroll 4
  for (int q = 0; q < ND / 4; ++q) {
    float4 v = xs[q];
    atomicAdd(&ar[q * 4 + 0], norm * v.x);
    atomicAdd(&ar[q * 4 + 1], norm * v.y);
    atomicAdd(&ar[q * 4 + 2], norm * v.z);
    atomicAdd(&ar[q * 4 + 3], norm * v.w);
  }
}

// per-feature sum / sum-of-squares over the batch axis (block partials + atomics)
__global__ void bn_stats_kernel(const float* __restrict__ h,
                                float* __restrict__ stats, int rows, int D) {
  __shared__ float red[2][4][ND];
  const int f = threadIdx.x;
  const int ry = threadIdx.y;
  float s = 0.f, s2 = 0.f;
  for (long long r = (long long)blockIdx.x * blockDim.y + ry; r < rows;
       r += (long long)gridDim.x * blockDim.y) {
    float v = h[r * D + f];
    s += v;
    s2 += v * v;
  }
  red[0][ry][f] = s;
  red[1][ry][f] = s2;
  __syncthreads();
  if (ry == 0) {
    for (int y = 1; y < (int)blockDim.y; ++y) {
      s += red[0][y][f];
      s2 += red[1][y][f];
    }
    atomicAdd(&stats[f], s);
    atomicAdd(&stats[D + f], s2);
  }
}

// in-place: out_ef = relu(BN(edge_feat)) + edge_attr
__global__ void apply_edge_kernel(float* __restrict__ out_ef,
                                  const float* __restrict__ edge_attr,
                                  const float* __restrict__ stats,
                                  const float* __restrict__ gamma,
                                  const float* __restrict__ beta) {
  long long idx = (long long)blockIdx.x * blockDim.x + threadIdx.x;
  if (idx >= (long long)N_EDGES * ED) return;
  const int f = (int)(idx % ED);
  const float inv = 1.f / (float)N_EDGES;
  const float mu = stats[f] * inv;
  const float var = stats[ED + f] * inv - mu * mu;
  float v = (out_ef[idx] - mu) * rsqrtf(var + BN_EPS) * gamma[f] + beta[f];
  out_ef[idx] = fmaxf(v, 0.f) + edge_attr[idx];
}

// out = relu(BN(agg)) + x      (bg cancels inside BN: constant per-feature shift)
__global__ void apply_node_kernel(const float* __restrict__ agg,
                                  const float* __restrict__ x,
                                  const float* __restrict__ stats,
                                  const float* __restrict__ gamma,
                                  const float* __restrict__ beta,
                                  float* __restrict__ out) {
  long long idx = (long long)blockIdx.x * blockDim.x + threadIdx.x;
  if (idx >= (long long)N_NODES * ND) return;
  const int f = (int)(idx % ND);
  const float inv = 1.f / (float)N_NODES;
  const float mu = stats[f] * inv;
  const float var = stats[ND + f] * inv - mu * mu;
  float v = (agg[idx] - mu) * rsqrtf(var + BN_EPS) * gamma[f] + beta[f];
  out[idx] = fmaxf(v, 0.f) + x[idx];
}

// ---------------------------------------------------------------------------
extern "C" void kernel_launch(void* const* d_in, const int* in_sizes, int n_in,
                              void* d_out, int out_size, void* d_ws,
                              size_t ws_size, hipStream_t stream) {
  const float* x          = (const float*)d_in[0];
  const long long* ei     = (const long long*)d_in[1];  // int64 edge_index
  const float* edge_f     = (const float*)d_in[2];
  const float* edge_attr  = (const float*)d_in[3];
  const float* Wn         = (const float*)d_in[4];
  const float* bn_        = (const float*)d_in[5];
  const float* Wec        = (const float*)d_in[6];
  const float* bec        = (const float*)d_in[7];
  const float* Ww         = (const float*)d_in[8];
  const float* bw         = (const float*)d_in[9];
  const float* Wg         = (const float*)d_in[10];
  // d_in[11] = bg: cancels inside batch-mean subtraction -> unused
  const float* gamma_e    = (const float*)d_in[12];
  const float* beta_e     = (const float*)d_in[13];
  const float* gamma_n    = (const float*)d_in[14];
  const float* beta_n     = (const float*)d_in[15];

  float* out_x  = (float*)d_out;                          // [N, 96]
  float* out_ef = (float*)d_out + (size_t)N_NODES * ND;   // [E, 64]

  char* ws = (char*)d_ws;
  size_t off = 0;
  auto carve = [&](size_t bytes) -> char* {
    char* p = ws + off;
    off = (off + bytes + 255) & ~(size_t)255;
    return p;
  };
  _Float16* hWn   = (_Float16*)carve((size_t)ND * ND * 2);
  _Float16* hWec  = (_Float16*)carve((size_t)ED * CATD * 2);
  _Float16* hWg   = (_Float16*)carve((size_t)ND * ND * 2);
  float* w_buf    = (float*)carve((size_t)N_EDGES * 4);
  float* deg      = (float*)carve((size_t)N_NODES * 4);
  float* dinv     = (float*)carve((size_t)N_NODES * 4);
  float* xw       = (float*)carve((size_t)N_NODES * ND * 4);
  float* agg      = (float*)carve((size_t)N_NODES * ND * 4);
  float* stats_e  = (float*)carve((size_t)2 * ED * 4);
  float* stats_n  = (float*)carve((size_t)2 * ND * 4);

  hipMemsetAsync(deg, 0, (size_t)N_NODES * 4, stream);
  hipMemsetAsync(agg, 0, (size_t)N_NODES * ND * 4, stream);
  hipMemsetAsync(stats_e, 0, (size_t)2 * ED * 4, stream);
  hipMemsetAsync(stats_n, 0, (size_t)2 * ND * 4, stream);

  convert_weights_kernel<<<(ED * CATD + 255) / 256, 256, 0, stream>>>(
      Wn, Wec, Wg, hWn, hWec, hWg);

  edge_kernel<<<N_EDGES / 16 / EDGE_WAVES, 256, 0, stream>>>(
      x, ei, edge_attr, edge_f, hWn, hWec, bn_, bec, Ww, bw, out_ef, w_buf,
      deg);

  node_gemm_kernel<<<(N_NODES / 16 + NG_WAVES - 1) / NG_WAVES, 256, 0,
                     stream>>>(x, hWg, xw);

  dinv_kernel<<<(N_NODES + 255) / 256, 256, 0, stream>>>(deg, dinv);

  scatter_kernel<<<(N_EDGES + 255) / 256, 256, 0, stream>>>(ei, w_buf, dinv,
                                                            xw, agg);

  bn_stats_kernel<<<dim3(512), dim3(ED, 4), 0, stream>>>(out_ef, stats_e,
                                                         N_EDGES, ED);
  bn_stats_kernel<<<dim3(512), dim3(ND, 2), 0, stream>>>(agg, stats_n,
                                                         N_NODES, ND);

  apply_edge_kernel<<<(int)(((long long)N_EDGES * ED + 255) / 256), 256, 0,
                      stream>>>(out_ef, edge_attr, stats_e, gamma_e, beta_e);
  apply_node_kernel<<<(int)(((long long)N_NODES * ND + 255) / 256), 256, 0,
                      stream>>>(agg, x, stats_n, gamma_n, beta_n, out_x);
}
